// Model_26371099198079
// MI455X (gfx1250) — matrix operand order repaired and checked by reading, not compiled, not run on hardware
//
#include <hip/hip_runtime.h>

typedef __attribute__((ext_vector_type(16))) __bf16 v16bf;
typedef __attribute__((ext_vector_type(8)))  float  v8f;

#define NBRS 32
#define DIMF 64
#define SF   68   // f32 row stride (pads banks; 272B keeps float4 alignment)
#define SH   68   // bf16 row stride (136B, u32-aligned for even k)

union FragBF {
    v16bf v;
    unsigned u[8];
};

__global__ void zero_f32(float* __restrict__ p, int n) {
    int i = blockIdx.x * blockDim.x + threadIdx.x;
    if (i < n) p[i] = 0.0f;
}

__global__ void deg_count(const int* __restrict__ nbr, float* __restrict__ deg, int n) {
    int i = blockIdx.x * blockDim.x + threadIdx.x;
    if (i < n) atomicAdd(&deg[nbr[i]], 1.0f);
}

__global__ void __launch_bounds__(32)
facloc_kernel(const float* __restrict__ node_f,
              const int*   __restrict__ nbr,
              const int*   __restrict__ cats,
              const float* __restrict__ deg,
              float*       __restrict__ out,
              int num_dst)
{
    __shared__ float  Xf [NBRS * SF];     // deg-scaled mail rows, f32
    __shared__ __bf16 Xhi[NBRS * SH];     // bf16 high part
    __shared__ __bf16 Xlo[NBRS * SH];     // bf16 residual
    __shared__ float  sims[NBRS * 33];    // dist, then exp(-d/mean)
    __shared__ float  sq_sh[NBRS];
    __shared__ float  cache_sh[NBRS];
    __shared__ int    idx_sh[NBRS];
    __shared__ float  sc_sh[NBRS];

    const int d = blockIdx.x;
    const int t    = threadIdx.x;   // 0..31, wave32
    const int hgrp = t >> 4;        // lane half
    const int lo16 = t & 15;

    // ---- per-lane neighbor bookkeeping ----
    int   src = nbr[d * NBRS + t];
    int   cat = cats[src];
    float sc  = rsqrtf(fmaxf(deg[src], 1.0f));
    idx_sh[t] = src;
    sc_sh[t]  = sc;
    bool fb = (__ballot(cat == -1) != 0ull);
    __syncthreads();

    // ---- cooperative gather: row r, each lane brings 2 features (256B/row) ----
    for (int r = 0; r < NBRS; ++r) {
        int   s     = idx_sh[r];
        float scale = sc_sh[r];
        float2 vv = *(reinterpret_cast<const float2*>(node_f + (size_t)s * DIMF) + t);
        float x0 = vv.x * scale;
        float x1 = vv.y * scale;
        int c0 = 2 * t;
        Xf[r * SF + c0]     = x0;
        Xf[r * SF + c0 + 1] = x1;
        __bf16 h0 = (__bf16)x0;
        __bf16 h1 = (__bf16)x1;
        Xhi[r * SH + c0]     = h0;
        Xhi[r * SH + c0 + 1] = h1;
        Xlo[r * SH + c0]     = (__bf16)(x0 - (float)h0);
        Xlo[r * SH + c0 + 1] = (__bf16)(x1 - (float)h1);
    }
    __syncthreads();

    // ---- f32 squared norms, one row per lane ----
    {
        float a = 0.0f;
        const float4* rp = reinterpret_cast<const float4*>(&Xf[t * SF]);
        #pragma unroll
        for (int j = 0; j < DIMF / 4; ++j) {
            float4 q = rp[j];
            a += q.x * q.x + q.y * q.y + q.z * q.z + q.w * q.w;
        }
        sq_sh[t] = a;
    }
    __syncthreads();

    // ---- Gram via bf16 hi/lo split WMMA: S = X X^T, four 16x16 tiles ----
    float dsum = 0.0f;
    #pragma unroll
    for (int a = 0; a < 2; ++a) {
        // A fragments (16x32 16-bit layout, ISA 7.12.2), both K-blocks
        FragBF aH[2], aL[2];
        {
            int row = a * 16 + lo16;   // M = lane & 15 for both halves
            #pragma unroll
            for (int kb = 0; kb < 2; ++kb) {
                #pragma unroll
                for (int v = 0; v < 8; ++v) {
                    int k = ((v >> 2) << 4) + hgrp * 8 + ((v & 3) << 1) + kb * 32;
                    aH[kb].u[v] = *reinterpret_cast<const unsigned*>(&Xhi[row * SH + k]);
                    aL[kb].u[v] = *reinterpret_cast<const unsigned*>(&Xlo[row * SH + k]);
                }
            }
        }
        #pragma unroll
        for (int b = 0; b < 2; ++b) {
            v8f acc = {};
            #pragma unroll
            for (int kb = 0; kb < 2; ++kb) {
                FragBF bH, bL;
                int row = b * 16 + lo16;   // B column = lane & 15
                #pragma unroll
                for (int v = 0; v < 8; ++v) {
                    int k = ((v >> 2) << 4) + hgrp * 8 + ((v & 3) << 1) + kb * 32;  // 32x16 16-bit B layout
                    bH.u[v] = *reinterpret_cast<const unsigned*>(&Xhi[row * SH + k]);
                    bL.u[v] = *reinterpret_cast<const unsigned*>(&Xlo[row * SH + k]);
                }
                acc = __builtin_amdgcn_wmma_f32_16x16x32_bf16(false, aH[kb].v, false, bH.v,
                                                              (short)0, acc, false, false);
                acc = __builtin_amdgcn_wmma_f32_16x16x32_bf16(false, aH[kb].v, false, bL.v,
                                                              (short)0, acc, false, false);
                acc = __builtin_amdgcn_wmma_f32_16x16x32_bf16(false, aL[kb].v, false, bH.v,
                                                              (short)0, acc, false, false);
            }
            // tile post-process: d2 = |xi|^2 + |xj|^2 - 2 xi.xj ; dist = sqrt(max(d2,0))
            float sqn = sq_sh[b * 16 + lo16];
            int   gn  = b * 16 + lo16;
            #pragma unroll
            for (int v = 0; v < 8; ++v) {
                int   gm  = a * 16 + v + 8 * hgrp;     // C/D layout: M = v + 8*(lane>=16)
                float dot = acc[v];
                float d2  = fmaxf(fmaf(-2.0f, dot, sq_sh[gm] + sqn), 0.0f);
                // raw v_sqrt_f32: d2 is O(1); feeds exp+argmax only
                float dst = (gm == gn) ? 0.0f : __builtin_amdgcn_sqrtf(d2);
                sims[gm * 33 + gn] = dst;
                dsum += dst;
            }
        }
    }

    // ---- mean over 32x32, then sims = exp(-dist/mean) ----
    #pragma unroll
    for (int off = 16; off > 0; off >>= 1)
        dsum += __shfl_xor(dsum, off, 32);
    // ninv = -1/mean via raw v_rcp_f32 (1 ulp, plenty for exp+argmax)
    float ninv = -__builtin_amdgcn_rcpf(dsum * (1.0f / (NBRS * NBRS)));   // SIGMA = 1
    __syncthreads();
    for (int r = 0; r < NBRS; ++r) {
        int off = r * 33 + t;
        sims[off] = __expf(sims[off] * ninv);
    }
    cache_sh[t] = 0.0f;
    __syncthreads();

    // ---- greedy facility-location, K = 8 ----
    float o0 = 0.0f, o1 = 0.0f;
    #pragma unroll 1
    for (int it = 0; it < 8; ++it) {
        float g = 0.0f;
        #pragma unroll
        for (int n = 0; n < NBRS; ++n)
            g += fmaxf(sims[t * 33 + n] - cache_sh[n], 0.0f);
        // wave argmax, first-max tie-break (matches jnp.argmax)
        float bg = g; int bj = t;
        #pragma unroll
        for (int off = 16; off > 0; off >>= 1) {
            float og = __shfl_xor(bg, off, 32);
            int   oj = __shfl_xor(bj, off, 32);
            if (og > bg || (og == bg && oj < bj)) { bg = og; bj = oj; }
        }
        int sel = bj;   // uniform across wave
        __syncthreads();
        cache_sh[t] = fmaxf(cache_sh[t], sims[sel * 33 + t]);
        o0 += Xf[sel * SF + t];
        o1 += Xf[sel * SF + t + 32];
        __syncthreads();
    }

    // ---- fallback branch: plain sum over all neighbors ----
    if (fb) {
        o0 = 0.0f; o1 = 0.0f;
        for (int r = 0; r < NBRS; ++r) {
            o0 += Xf[r * SF + t];
            o1 += Xf[r * SF + t + 32];
        }
    }

    const float inorm = 0.17677669529663687f;   // 32^-0.5 in-degree norm
    out[(size_t)d * DIMF + t]      = o0 * inorm;
    out[(size_t)d * DIMF + t + 32] = o1 * inorm;
}

extern "C" void kernel_launch(void* const* d_in, const int* in_sizes, int n_in,
                              void* d_out, int out_size, void* d_ws, size_t ws_size,
                              hipStream_t stream) {
    const float* node_f = (const float*)d_in[0];   // [60000, 64] f32
    const int*   nbr    = (const int*)d_in[1];     // [50000, 32] i32
    const int*   cats   = (const int*)d_in[2];     // [60000]     i32
    float*       out    = (float*)d_out;           // [50000, 64] f32
    float*       deg    = (float*)d_ws;            // [num_src]   f32 scratch

    int num_src   = in_sizes[0] / DIMF;            // 60000
    int num_edges = in_sizes[1];                   // 50000 * 32
    int num_dst   = num_edges / NBRS;              // 50000

    zero_f32 <<<(num_src   + 255) / 256, 256, 0, stream>>>(deg, num_src);
    deg_count<<<(num_edges + 255) / 256, 256, 0, stream>>>(nbr, deg, num_edges);
    facloc_kernel<<<num_dst, 32, 0, stream>>>(node_f, nbr, cats, deg, out, num_dst);
}
